// MultiHeadAttention_65481071395029
// MI455X (gfx1250) — matrix operand order repaired and checked by reading, not compiled, not run on hardware
//
#include <hip/hip_runtime.h>
#include <hip/hip_bf16.h>

typedef __attribute__((ext_vector_type(16))) __bf16 v16bf;
typedef __attribute__((ext_vector_type(8)))  float  v8f;

union FragAB { uint4 u[2]; v16bf v; };

__device__ __forceinline__ unsigned short f2bf(float x) {
    unsigned int u = __float_as_uint(x);
    unsigned int r = (u + 0x7fffu + ((u >> 16) & 1u)) >> 16;
    return (unsigned short)r;
}
__device__ __forceinline__ float bf2f(unsigned short h) {
    return __uint_as_float(((unsigned int)h) << 16);
}

// ---------------------------------------------------------------------------
// Kernel 1: projection GEMM  out[m,n] = sum_k X[m,k] * W[n,k] + bias[n]
// M=4096 (b*S+s), N=1024 (h*DK+dk), K=1024. Output stored bf16 as [B,H,S,DK].
// Block: 64x64 tile, 256 threads (8 waves), K-step 32, WMMA bf16.
// ---------------------------------------------------------------------------
__global__ __launch_bounds__(256) void proj_gemm(
    const float* __restrict__ X, const float* __restrict__ W,
    const float* __restrict__ bias, unsigned short* __restrict__ out)
{
    constexpr int K = 1024, S = 1024, DK = 64, H = 16;
    constexpr int LDA = 40;  // padded row stride (halfwords) for 32-wide K tile
    __shared__ __align__(16) unsigned short As[64 * LDA];
    __shared__ __align__(16) unsigned short Bs[64 * LDA];

    const int m0 = blockIdx.x * 64;
    const int n0 = blockIdx.y * 64;
    const int t  = threadIdx.x;
    const int w  = t >> 5;
    const int l  = t & 31;
    const int lm = l & 15;
    const int half = l >> 4;
    const int wm = (w & 3) * 16;   // wave M offset in block
    const int wn = (w >> 2) * 32;  // wave N offset in block

    const int lrow = t >> 2;        // 0..63
    const int lcol = (t & 3) * 8;   // 0,8,16,24

    v8f acc[2];
    acc[0] = v8f{};
    acc[1] = v8f{};

    const float* sa = X + (size_t)(m0 + lrow) * K + lcol;
    const float* sb = W + (size_t)(n0 + lrow) * K + lcol;

    for (int k0 = 0; k0 < K; k0 += 32) {
        // stage A (64x32) and B (64x32) tiles, converting f32 -> bf16
        {
            unsigned short* da = As + lrow * LDA + lcol;
            unsigned short* db = Bs + lrow * LDA + lcol;
#pragma unroll
            for (int i = 0; i < 8; ++i) da[i] = f2bf(sa[k0 + i]);
#pragma unroll
            for (int i = 0; i < 8; ++i) db[i] = f2bf(sb[k0 + i]);
            // prefetch next K slab (global_prefetch_b8) while WMMAs run
            if (k0 + 32 < K) {
                __builtin_prefetch(sa + k0 + 32, 0, 3);
                __builtin_prefetch(sb + k0 + 32, 0, 3);
            }
        }
        __syncthreads();

        // A fragment (16x32): lane half picks K groups {0..7,16..23} / {8..15,24..31}
        FragAB a, b0, b1;
        const unsigned short* ap = As + (wm + lm) * LDA;
        a.u[0] = *(const uint4*)(ap + half * 8);
        a.u[1] = *(const uint4*)(ap + 16 + half * 8);
        // B fragments (32x16 each): lane half picks K groups {0..7,16..23} / {8..15,24..31}
        const unsigned short* bp0 = Bs + (wn + lm) * LDA + half * 8;
        b0.u[0] = *(const uint4*)(bp0);
        b0.u[1] = *(const uint4*)(bp0 + 16);
        const unsigned short* bp1 = Bs + (wn + 16 + lm) * LDA + half * 8;
        b1.u[0] = *(const uint4*)(bp1);
        b1.u[1] = *(const uint4*)(bp1 + 16);

        acc[0] = __builtin_amdgcn_wmma_f32_16x16x32_bf16(
            false, a.v, false, b0.v, (short)0, acc[0], false, false);
        acc[1] = __builtin_amdgcn_wmma_f32_16x16x32_bf16(
            false, a.v, false, b1.v, (short)0, acc[1], false, false);
        __syncthreads();
    }

    // epilogue: bias add, convert to bf16, scatter into [B,H,S,DK]
#pragma unroll
    for (int tc = 0; tc < 2; ++tc) {
        const int n  = n0 + wn + tc * 16 + lm;
        const float bv = bias[n];
        const int h = n >> 6, dk = n & 63;
#pragma unroll
        for (int r = 0; r < 8; ++r) {
            const int m = m0 + wm + half * 8 + r;
            const int b = m >> 10, s = m & 1023;
            out[(((size_t)(b * H + h) * S) + s) * DK + dk] = f2bf(acc[tc][r] + bv);
        }
    }
}

// ---------------------------------------------------------------------------
// Kernel 2: aw[b,h,d2] = sum_c (aspect[b]·Wd[c] + bd[c]) * weight_m[h,c,d2]
// ---------------------------------------------------------------------------
__global__ __launch_bounds__(256) void aw_kernel(
    const float* __restrict__ aspect, const float* __restrict__ Wd,
    const float* __restrict__ bd, const float* __restrict__ weight_m,
    float* __restrict__ aw)
{
    __shared__ float av[1024];
    __shared__ float asp[64];
    const int b = blockIdx.x, t = threadIdx.x;
    for (int i = t; i < 1024; i += 256) av[i] = aspect[b * 1024 + i];
    __syncthreads();
    if (t < 64) {
        float s = 0.f;
        const float* wr = Wd + t * 1024;
        for (int i = 0; i < 1024; ++i) s += av[i] * wr[i];
        asp[t] = s + bd[t];
    }
    __syncthreads();
    for (int o = t; o < 1024; o += 256) {
        const int h = o >> 6, d2 = o & 63;
        float s = 0.f;
        const float* wm = weight_m + h * 4096 + d2;
#pragma unroll 8
        for (int c = 0; c < 64; ++c) s += asp[c] * wm[c * 64];
        aw[b * 1024 + o] = s;
    }
}

// ---------------------------------------------------------------------------
// Kernel 3: aspsc[b,h,s] = tanh( aw[b,h]·khat[b,h,s,:] + bias_m )
// ---------------------------------------------------------------------------
__global__ __launch_bounds__(256) void aspscore_kernel(
    const float* __restrict__ aw, const unsigned short* __restrict__ khat,
    const float* __restrict__ bias_m, float* __restrict__ aspsc)
{
    const int idx = blockIdx.x * 256 + threadIdx.x;   // over B*H*S
    const int bh = idx >> 10;
    const unsigned short* kr = khat + (size_t)idx * 64;
    const float* a = aw + bh * 64;
    float s = 0.f;
#pragma unroll 8
    for (int i = 0; i < 64; ++i) s += a[i] * bf2f(kr[i]);
    aspsc[idx] = tanhf(s + bias_m[0]);
}

// ---------------------------------------------------------------------------
// Kernel 4: fused scores + mask + softmax.
// Block = (b, h, 16-row q tile). 8 waves x 8 column tiles x WMMA(16x16x32 x2).
// Scores stay in registers (64 f32/lane); two-level online-softmax reduction.
// ---------------------------------------------------------------------------
__global__ __launch_bounds__(256) void attn_kernel(
    const unsigned short* __restrict__ qhat, const unsigned short* __restrict__ khat,
    const float* __restrict__ aspsc, const float* __restrict__ shortb,
    const int* __restrict__ mask, float* __restrict__ out)
{
    constexpr int S = 1024, DK = 64;
    constexpr int LQ = 72;  // padded q-tile row stride (halfwords)
    const int blk = blockIdx.x;
    const int qt = blk & 63;
    const int bh = blk >> 6;
    const int b  = bh >> 4;
    const int t = threadIdx.x, w = t >> 5, l = t & 31;
    const int lm = l & 15, half = l >> 4;

    __shared__ __align__(16) unsigned short Qs[16 * LQ];
    __shared__ float partM[16][8];
    __shared__ float partS[16][8];
    __shared__ float finM[16], finS[16];

    // stage q tile (16 rows x 64 dk, bf16)
    {
        const int row = t >> 4;
        const int dk4 = (t & 15) * 4;
        const unsigned short* src =
            qhat + ((size_t)bh * S + qt * 16 + row) * DK + dk4;
        *(uint2*)(Qs + row * LQ + dk4) = *(const uint2*)src;
    }
    __syncthreads();

    // A fragments for K(dk)=0..31 and 32..63
    FragAB a0, a1;
    const unsigned short* ap = Qs + lm * LQ;
    a0.u[0] = *(const uint4*)(ap + half * 8);
    a0.u[1] = *(const uint4*)(ap + 16 + half * 8);
    a1.u[0] = *(const uint4*)(ap + 32 + half * 8);
    a1.u[1] = *(const uint4*)(ap + 48 + half * 8);

    const unsigned short* kbase = khat + (size_t)bh * S * DK;
    const float* shbase = shortb + (size_t)bh * S * S + (size_t)(qt * 16) * S;
    const int*   mbase  = mask   + (size_t)b  * S * S + (size_t)(qt * 16) * S;

    // warm the HBM pipe for the first column tile of the short/mask streams
    {
        const int col0 = (w * 8) * 16 + lm;
        __builtin_prefetch(shbase + (half * 8) * S + col0, 0, 3);
        __builtin_prefetch(mbase + (half * 8) * S + col0, 0, 3);
    }

    float xs[8][8];
    float rm[8], rs[8];
#pragma unroll
    for (int r = 0; r < 8; ++r) { rm[r] = -3.0e38f; rs[r] = 0.f; }

#pragma unroll
    for (int nt = 0; nt < 8; ++nt) {
        const int col = (w * 8 + nt) * 16 + lm;
        FragAB b0, b1;
        const unsigned short* kp = kbase + (size_t)col * DK + half * 8;
        b0.u[0] = *(const uint4*)(kp);
        b0.u[1] = *(const uint4*)(kp + 16);
        b1.u[0] = *(const uint4*)(kp + 32);
        b1.u[1] = *(const uint4*)(kp + 48);

        v8f c = v8f{};
        c = __builtin_amdgcn_wmma_f32_16x16x32_bf16(
            false, a0.v, false, b0.v, (short)0, c, false, false);
        c = __builtin_amdgcn_wmma_f32_16x16x32_bf16(
            false, a1.v, false, b1.v, (short)0, c, false, false);

        // prefetch next tile's short/mask cachelines while WMMAs drain
        if (nt < 7) {
            const int ncol = col + 16;
            __builtin_prefetch(shbase + (half * 8) * S + ncol, 0, 3);
            __builtin_prefetch(shbase + (half * 8 + 4) * S + ncol, 0, 3);
            __builtin_prefetch(mbase + (half * 8) * S + ncol, 0, 3);
        }

        const float aspv = aspsc[bh * S + col];
#pragma unroll
        for (int r = 0; r < 8; ++r) {
            const int row = half * 8 + r;
            float x = c[r] * 0.125f + aspv + shbase[row * S + col];
            x = (mbase[row * S + col] == 0) ? -1.0e9f : x;
            xs[nt][r] = x;
            rm[r] = fmaxf(rm[r], x);
        }
    }

    // per-wave row max across 16 lanes (each half-wave owns rows 0-7 / 8-15)
#pragma unroll
    for (int r = 0; r < 8; ++r) {
        float m = rm[r];
        for (int off = 8; off; off >>= 1) m = fmaxf(m, __shfl_xor(m, off, 16));
        rm[r] = m;
    }
#pragma unroll
    for (int nt = 0; nt < 8; ++nt)
#pragma unroll
        for (int r = 0; r < 8; ++r) rs[r] += __expf(xs[nt][r] - rm[r]);
#pragma unroll
    for (int r = 0; r < 8; ++r) {
        float sv = rs[r];
        for (int off = 8; off; off >>= 1) sv += __shfl_xor(sv, off, 16);
        rs[r] = sv;
    }
    if (lm == 0) {
#pragma unroll
        for (int r = 0; r < 8; ++r) {
            partM[half * 8 + r][w] = rm[r];
            partS[half * 8 + r][w] = rs[r];
        }
    }
    __syncthreads();
    if (t < 16) {
        float M = -3.0e38f;
#pragma unroll
        for (int i = 0; i < 8; ++i) M = fmaxf(M, partM[t][i]);
        float Ssum = 0.f;
#pragma unroll
        for (int i = 0; i < 8; ++i) Ssum += partS[t][i] * __expf(partM[t][i] - M);
        finM[t] = M;
        finS[t] = 1.0f / Ssum;
    }
    __syncthreads();

    float fM[8], fS[8];
#pragma unroll
    for (int r = 0; r < 8; ++r) { fM[r] = finM[half * 8 + r]; fS[r] = finS[half * 8 + r]; }

    float* obase = out + (size_t)bh * S * S + (size_t)(qt * 16) * S;
#pragma unroll
    for (int nt = 0; nt < 8; ++nt) {
        const int col = (w * 8 + nt) * 16 + lm;
#pragma unroll
        for (int r = 0; r < 8; ++r) {
            const int row = half * 8 + r;
            obase[row * S + col] = __expf(xs[nt][r] - fM[r]) * fS[r];
        }
    }
}

// ---------------------------------------------------------------------------
extern "C" void kernel_launch(void* const* d_in, const int* in_sizes, int n_in,
                              void* d_out, int out_size, void* d_ws, size_t ws_size,
                              hipStream_t stream) {
    const float* query  = (const float*)d_in[0];
    const float* key    = (const float*)d_in[1];
    const int*   maskp  = (const int*)d_in[2];
    const float* aspect = (const float*)d_in[3];
    const float* shortb = (const float*)d_in[4];
    const float* Wq = (const float*)d_in[5];
    const float* bq = (const float*)d_in[6];
    const float* Wk = (const float*)d_in[7];
    const float* bk = (const float*)d_in[8];
    const float* Wd = (const float*)d_in[9];
    const float* bd = (const float*)d_in[10];
    const float* weight_m = (const float*)d_in[11];
    const float* bias_m   = (const float*)d_in[12];
    float* out = (float*)d_out;

    char* ws = (char*)d_ws;
    unsigned short* qhat = (unsigned short*)ws;                             // 8 MiB
    unsigned short* khat = (unsigned short*)(ws + (size_t)8 * 1024 * 1024); // 8 MiB
    float* aw    = (float*)(ws + (size_t)16 * 1024 * 1024);                 // 16 KiB
    float* aspsc = (float*)(ws + (size_t)16 * 1024 * 1024 + 64 * 1024);     // 256 KiB

    // Q and K projections (M=4096, N=1024, K=1024)
    proj_gemm<<<dim3(64, 16), 256, 0, stream>>>(query, Wq, bq, qhat);
    proj_gemm<<<dim3(64, 16), 256, 0, stream>>>(key, Wk, bk, khat);
    // aspect path
    aw_kernel<<<4, 256, 0, stream>>>(aspect, Wd, bd, weight_m, aw);
    aspscore_kernel<<<256, 256, 0, stream>>>(aw, khat, bias_m, aspsc);
    // fused scores + mask + softmax
    attn_kernel<<<4096, 256, 0, stream>>>(qhat, khat, aspsc, shortb, maskp, out);
}
